// NeRFMoETorch_81990925680843
// MI455X (gfx1250) — matrix operand order repaired and checked by reading, not compiled, not run on hardware
//
#include <hip/hip_runtime.h>

// ---------------- Problem constants (match reference) ----------------
#define NPTS 131072
#define DD   256
#define EE   8
#define CAPC 20480
#define PX   12
#define PD   4

typedef __attribute__((ext_vector_type(16))) __bf16 v16bf;
typedef __attribute__((ext_vector_type(8)))  float  v8f;

// ---------------- bf16 helpers (raw ushort storage) ----------------
__device__ __forceinline__ unsigned short f2bf(float f) {
    union { float f; unsigned int u; } v; v.f = f;
    unsigned int r = v.u + 0x7FFFu + ((v.u >> 16) & 1u);   // round-to-nearest-even
    return (unsigned short)(r >> 16);
}
__device__ __forceinline__ float bf2f(unsigned short h) {
    union { unsigned int u; float f; } v; v.u = ((unsigned int)h) << 16;
    return v.f;
}
__device__ __forceinline__ float wred(float v) {
    #pragma unroll
    for (int m = 16; m > 0; m >>= 1) v += __shfl_xor(v, m, 32);
    return v;
}

// ---------------- Weight repack: f32 [K x M] row-major -> WMMA B-fragment order ----------------
// Wp[((kt*CT+ct)*32 + lane)*16 + i] = bf16(W[kt*32 + (lane>>4)*16 + i][ct*16 + (lane&15)]),
// zero-padded for k >= K.  blockIdx.y = batch (expert) index.
__global__ void k_pack(const float* __restrict__ W, unsigned short* __restrict__ Wp,
                       int K, int Kp, int M) {
    long b = blockIdx.y;
    W  += b * (long)K * M;
    Wp += b * (long)Kp * M;
    long idx = (long)blockIdx.x * 256 + threadIdx.x;
    long total = (long)Kp * M;
    if (idx >= total) return;
    int i    = (int)(idx & 15);
    int lane = (int)((idx >> 4) & 31);
    long r   = idx >> 9;
    int CT   = M >> 4;
    int ct   = (int)(r % CT);
    int kt   = (int)(r / CT);
    int k = kt * 32 + (lane >> 4) * 16 + i;
    int m = ct * 16 + (lane & 15);
    float v = (k < K) ? W[(long)k * M + m] : 0.f;
    Wp[idx] = f2bf(v);
}

// ---------------- Generic bf16 WMMA GEMM ----------------
// C[n x M] (bf16, leading dim ldC) = act(A[n x K] @ W + bias), A split at K1 between A/A2p.
// Weights pre-packed (k_pack). Per-expert weight/bias strides via cap_blocks (row blocks per expert).
// Block: 256 thr = 8 waves; block tile 32 rows x M cols; wave tile 16 rows x (16*TPW) cols.
template <int TPW>
__global__ __launch_bounds__(256)
void k_gemm(const unsigned short* __restrict__ A, const unsigned short* __restrict__ A2p, int K1,
            const unsigned short* __restrict__ Wp, long wstride, int cap_blocks,
            const float* __restrict__ bias, long bstride,
            unsigned short* __restrict__ C, int ldC, int K, int M, int relu) {
    __shared__ unsigned short As[32][520];          // 520*2 bytes/row keeps 16B alignment
    const int tid  = threadIdx.x;
    const int wave = tid >> 5;
    const int lane = tid & 31;
    const long row0 = (long)blockIdx.x * 32;
    const int expert = cap_blocks ? ((int)blockIdx.x / cap_blocks) : 0;

    // ---- stage 32 x K A-tile into LDS (16B vector copies) ----
    const int Kv = K >> 3;                           // 8-half segments per row
    for (int u = tid; u < 32 * Kv; u += 256) {
        int r  = u / Kv;
        int k0 = (u - r * Kv) << 3;
        long grow = row0 + r;
        const unsigned short* src = (k0 < K1)
            ? (A   + grow * (long)K1       + k0)
            : (A2p + grow * (long)(K - K1) + (k0 - K1));
        *(uint4*)&As[r][k0] = *(const uint4*)src;
    }
    __syncthreads();

    const int wr = wave & 1;                 // row tile within block (0..1)
    const int wcg = wave >> 1;               // column group (0..3)
    const int lo = lane & 15;
    const int hi = lane >> 4;
    const int hi8 = hi << 3;
    const int CT = M >> 4;
    const int arow = wr * 16 + lo;

    v8f acc[TPW];
    #pragma unroll
    for (int t = 0; t < TPW; ++t) acc[t] = (v8f){0.f,0.f,0.f,0.f,0.f,0.f,0.f,0.f};

    const uint4* wq = (const uint4*)(Wp + (long)expert * wstride);
    const int KT = K >> 5;
    for (int kt = 0; kt < KT; ++kt) {
        union { uint4 u[2]; v16bf v; } af;
        const int kb = kt << 5;
        af.u[0] = *(const uint4*)&As[arow][kb + (hi << 4)];
        af.u[1] = *(const uint4*)&As[arow][kb + (hi << 4) + 8];
        #pragma unroll
        for (int t = 0; t < TPW; ++t) {
            int ct = wcg * TPW + t;
            long bidx = ((long)(kt * CT + ct) * 32 + lane) * 2;   // uint4 units
            union { uint4 u[2]; v16bf v; } bf;
            bf.u[0] = wq[bidx];
            bf.u[1] = wq[bidx + 1];
            acc[t] = __builtin_amdgcn_wmma_f32_16x16x32_bf16(
                false, af.v, false, bf.v, (short)0, acc[t], false, false);
        }
    }

    // ---- epilogue: bias + optional relu, scatter per C/D layout ----
    const float* bp = bias + (long)expert * bstride;
    #pragma unroll
    for (int t = 0; t < TPW; ++t) {
        int ct  = wcg * TPW + t;
        int col = ct * 16 + lo;
        float bv = bp[col];
        #pragma unroll
        for (int r = 0; r < 8; ++r) {
            long row = row0 + wr * 16 + hi8 + r;
            float y = acc[t][r] + bv;
            if (relu) y = fmaxf(y, 0.f);
            C[row * (long)ldC + col] = f2bf(y);
        }
    }
}

// ---------------- posenc(xyz, 12 freqs) -> A0 [N x 96] bf16 (padded) ----------------
__global__ void k_posenc(const float* __restrict__ x, unsigned short* __restrict__ A0) {
    int t = blockIdx.x * 256 + threadIdx.x;
    if (t >= NPTS) return;
    const float* xr = x + (long)t * 7;
    unsigned short* a = A0 + (long)t * 96;
    float p[3] = {xr[0], xr[1], xr[2]};
    a[0] = f2bf(p[0]); a[1] = f2bf(p[1]); a[2] = f2bf(p[2]);
    float f = 1.f;
    for (int k = 0; k < PX; ++k) {
        #pragma unroll
        for (int d = 0; d < 3; ++d) {
            a[3 + k*6 + d]     = f2bf(__sinf(p[d] * f));
            a[3 + k*6 + 3 + d] = f2bf(__cosf(p[d] * f));
        }
        f *= 2.f;
    }
    for (int c = 75; c < 96; ++c) a[c] = 0;
}

// ---------------- Gate: layernorm(ext) @ wg -> softmax/argmax (one wave/token) ----------------
__global__ void k_gate(const unsigned short* __restrict__ ext,
                       const float* __restrict__ ln_g, const float* __restrict__ ln_b,
                       const float* __restrict__ wg,
                       int* __restrict__ eidx, float* __restrict__ gate) {
    int wave = threadIdx.x >> 5, lane = threadIdx.x & 31;
    long t = (long)blockIdx.x * 8 + wave;
    const unsigned short* row = ext + t * DD;
    uint4 u = *(const uint4*)(row + lane * 8);
    const unsigned short* hs = (const unsigned short*)&u;
    float xv[8], s = 0.f, s2 = 0.f;
    #pragma unroll
    for (int j = 0; j < 8; ++j) { xv[j] = bf2f(hs[j]); s += xv[j]; s2 += xv[j]*xv[j]; }
    s = wred(s); s2 = wred(s2);
    float m = s * (1.f / DD);
    float var = s2 * (1.f / DD) - m * m;
    float inv = rsqrtf(var + 1e-5f);
    float p[EE];
    #pragma unroll
    for (int e = 0; e < EE; ++e) p[e] = 0.f;
    #pragma unroll
    for (int j = 0; j < 8; ++j) {
        int c = lane * 8 + j;
        float ln = (xv[j] - m) * inv * ln_g[c] + ln_b[c];
        #pragma unroll
        for (int e = 0; e < EE; ++e) p[e] += ln * wg[c * EE + e];
    }
    #pragma unroll
    for (int e = 0; e < EE; ++e) p[e] = wred(p[e]);
    if (lane == 0) {
        float mx = p[0]; int ei = 0;
        #pragma unroll
        for (int e = 1; e < EE; ++e) if (p[e] > mx) { mx = p[e]; ei = e; }  // first-max, like argmax
        float se = 0.f;
        #pragma unroll
        for (int e = 0; e < EE; ++e) se += expf(p[e] - mx);
        eidx[t] = ei;
        gate[t] = 1.f / se;                      // softmax at argmax
    }
}

// ---------------- token-order capacity assignment (cumsum semantics) ----------------
__global__ void k_count(const int* __restrict__ eidx, int* __restrict__ blockCounts) {
    __shared__ int sc[256][EE];
    int tid = threadIdx.x, b = blockIdx.x;
    long base = (long)b * 4096 + tid * 16;
    int c[EE];
    #pragma unroll
    for (int e = 0; e < EE; ++e) c[e] = 0;
    for (int i = 0; i < 16; ++i) c[eidx[base + i]]++;
    #pragma unroll
    for (int e = 0; e < EE; ++e) sc[tid][e] = c[e];
    __syncthreads();
    for (int off = 1; off < 256; off <<= 1) {
        int v[EE];
        if (tid >= off) {
            #pragma unroll
            for (int e = 0; e < EE; ++e) v[e] = sc[tid - off][e];
        }
        __syncthreads();
        if (tid >= off) {
            #pragma unroll
            for (int e = 0; e < EE; ++e) sc[tid][e] += v[e];
        }
        __syncthreads();
    }
    if (tid == 255) {
        #pragma unroll
        for (int e = 0; e < EE; ++e) blockCounts[b * EE + e] = sc[255][e];
    }
}
__global__ void k_scan(const int* __restrict__ blockCounts, int* __restrict__ bases) {
    int e = threadIdx.x;
    if (e < EE) {
        int run = 0;
        for (int b = 0; b < 32; ++b) { bases[b * EE + e] = run; run += blockCounts[b * EE + e]; }
    }
}
__global__ void k_assign(const int* __restrict__ eidx, const int* __restrict__ bases,
                         int* __restrict__ loc) {
    __shared__ int sc[256][EE];
    int tid = threadIdx.x, b = blockIdx.x;
    long base = (long)b * 4096 + tid * 16;
    int c[EE];
    #pragma unroll
    for (int e = 0; e < EE; ++e) c[e] = 0;
    for (int i = 0; i < 16; ++i) c[eidx[base + i]]++;
    #pragma unroll
    for (int e = 0; e < EE; ++e) sc[tid][e] = c[e];
    __syncthreads();
    for (int off = 1; off < 256; off <<= 1) {
        int v[EE];
        if (tid >= off) {
            #pragma unroll
            for (int e = 0; e < EE; ++e) v[e] = sc[tid - off][e];
        }
        __syncthreads();
        if (tid >= off) {
            #pragma unroll
            for (int e = 0; e < EE; ++e) sc[tid][e] += v[e];
        }
        __syncthreads();
    }
    int my[EE];
    #pragma unroll
    for (int e = 0; e < EE; ++e) my[e] = bases[b * EE + e] + (tid ? sc[tid - 1][e] : 0);
    for (int i = 0; i < 16; ++i) { int e = eidx[base + i]; loc[base + i] = my[e]++; }
}

// ---------------- zero / scatter / combine ----------------
__global__ void k_zero(uint4* __restrict__ p, long n4) {
    long i = (long)blockIdx.x * blockDim.x + threadIdx.x;
    long stride = (long)gridDim.x * blockDim.x;
    uint4 z; z.x = z.y = z.z = z.w = 0u;
    for (; i < n4; i += stride) p[i] = z;
}
__global__ void k_scatter(const unsigned short* __restrict__ hB, const int* __restrict__ eidx,
                          const int* __restrict__ loc, unsigned short* __restrict__ X0) {
    int wave = threadIdx.x >> 5, lane = threadIdx.x & 31;
    long t = (long)blockIdx.x * 8 + wave;
    int l = loc[t];
    if (l >= CAPC) return;
    long dst = ((long)eidx[t] * CAPC + l) * DD;
    *(uint4*)(X0 + dst + lane * 8) = *(const uint4*)(hB + t * DD + lane * 8);
}
__global__ void k_combine(const unsigned short* __restrict__ X1, const int* __restrict__ eidx,
                          const int* __restrict__ loc, const float* __restrict__ gate,
                          const float* __restrict__ wsv, const float* __restrict__ bs,
                          unsigned short* __restrict__ hout, float* __restrict__ out) {
    int wave = threadIdx.x >> 5, lane = threadIdx.x & 31;
    long t = (long)blockIdx.x * 8 + wave;
    int l = loc[t];
    int keep = l < CAPC;
    int lc = keep ? l : (CAPC - 1);
    const unsigned short* row = X1 + ((long)eidx[t] * CAPC + lc) * DD;
    uint4 u = *(const uint4*)(row + lane * 8);
    const unsigned short* hs = (const unsigned short*)&u;
    float g = gate[t];
    unsigned short ob[8];
    float part = 0.f;
    #pragma unroll
    for (int j = 0; j < 8; ++j) {
        float v = keep ? bf2f(hs[j]) * g : 0.f;
        v = fmaxf(v, 0.f);
        ob[j] = f2bf(v);
        part += v * wsv[lane * 8 + j];
    }
    *(uint4*)(hout + t * DD + lane * 8) = *(uint4*)ob;
    part = wred(part);
    if (lane == 0) {
        float z = part + bs[0];
        out[t * 4 + 3] = (z > 20.f) ? z : log1pf(expf(z));   // softplus
    }
}

// ---------------- A2 tail: dir posenc + appearance embedding + zero pad ----------------
__global__ void k_tail(const float* __restrict__ x, const float* __restrict__ emb,
                       unsigned short* __restrict__ A2) {
    int t = blockIdx.x * 256 + threadIdx.x;
    if (t >= NPTS) return;
    const float* xr = x + (long)t * 7;
    unsigned short* a = A2 + (long)t * 352 + 256;
    float d[3] = {xr[3], xr[4], xr[5]};
    a[0] = f2bf(d[0]); a[1] = f2bf(d[1]); a[2] = f2bf(d[2]);
    float f = 1.f;
    for (int k = 0; k < PD; ++k) {
        #pragma unroll
        for (int dd = 0; dd < 3; ++dd) {
            a[3 + k*6 + dd]     = f2bf(__sinf(d[dd] * f));
            a[3 + k*6 + 3 + dd] = f2bf(__cosf(d[dd] * f));
        }
        f *= 2.f;
    }
    int ai = (int)xr[6];
    const float* er = emb + (long)ai * 48;
    for (int j = 0; j < 48; ++j) a[27 + j] = f2bf(er[j]);
    for (int j = 75; j < 96; ++j) a[j] = 0;
}

// ---------------- rgb head: sigmoid(h2 @ wc + bc) (one wave/token) ----------------
__global__ void k_rgb(const unsigned short* __restrict__ h2, const float* __restrict__ wcm,
                      const float* __restrict__ bc, float* __restrict__ out) {
    int wave = threadIdx.x >> 5, lane = threadIdx.x & 31;
    long t = (long)blockIdx.x * 8 + wave;
    const unsigned short* row = h2 + t * 128;
    uint2 u = *(const uint2*)(row + lane * 4);
    const unsigned short* hs = (const unsigned short*)&u;
    float p[3] = {0.f, 0.f, 0.f};
    #pragma unroll
    for (int j = 0; j < 4; ++j) {
        float v = bf2f(hs[j]);
        int c = lane * 4 + j;
        #pragma unroll
        for (int e = 0; e < 3; ++e) p[e] += v * wcm[c * 3 + e];
    }
    #pragma unroll
    for (int e = 0; e < 3; ++e) p[e] = wred(p[e]);
    if (lane == 0) {
        #pragma unroll
        for (int e = 0; e < 3; ++e) out[t * 4 + e] = 1.f / (1.f + expf(-(p[e] + bc[e])));
    }
}

// =====================================================================
extern "C" void kernel_launch(void* const* d_in, const int* in_sizes, int n_in,
                              void* d_out, int out_size, void* d_ws, size_t ws_size,
                              hipStream_t stream) {
    (void)in_sizes; (void)n_in; (void)out_size; (void)ws_size;
    const float* x     = (const float*)d_in[0];
    const float* emb_a = (const float*)d_in[1];
    const float* w_xyz = (const float*)d_in[2];
    const float* b_xyz = (const float*)d_in[3];
    const float* wg1   = (const float*)d_in[4];
    const float* bg1   = (const float*)d_in[5];
    const float* wg2   = (const float*)d_in[6];
    const float* bg2   = (const float*)d_in[7];
    const float* ln_g  = (const float*)d_in[8];
    const float* ln_b  = (const float*)d_in[9];
    const float* wg    = (const float*)d_in[10];
    const float* eW_a  = (const float*)d_in[11];
    const float* eb_a  = (const float*)d_in[12];
    const float* eW3   = (const float*)d_in[13];
    const float* eb3   = (const float*)d_in[14];
    const float* eW_b  = (const float*)d_in[15];
    const float* eb_b  = (const float*)d_in[16];
    const float* w1    = (const float*)d_in[17];
    const float* b1    = (const float*)d_in[18];
    const float* w2    = (const float*)d_in[19];
    const float* b2    = (const float*)d_in[20];
    const float* wsv   = (const float*)d_in[21];
    const float* bs    = (const float*)d_in[22];
    const float* wcm   = (const float*)d_in[23];
    const float* bc    = (const float*)d_in[24];
    float* out = (float*)d_out;

    // ---- workspace bump allocator (byte offsets, 256B aligned) ----
    char* base = (char*)d_ws;
    size_t off = 0;
    auto alloc = [&](size_t bytes) { char* p = base + off; off = (off + bytes + 255) & ~(size_t)255; return p; };

    unsigned short* Wp0 = (unsigned short*)alloc((size_t)96  * 256 * 2);          // w_xyz
    unsigned short* Wp1 = (unsigned short*)alloc((size_t)256 * 256 * 2);          // wg1
    unsigned short* Wp2 = (unsigned short*)alloc((size_t)256 * 256 * 2);          // wg2
    unsigned short* Wp3 = (unsigned short*)alloc((size_t)256 * 256 * 2);          // w1
    unsigned short* Wp4 = (unsigned short*)alloc((size_t)352 * 128 * 2);          // w2
    unsigned short* Wp5 = (unsigned short*)alloc((size_t)24 * 256 * 256 * 2);     // eW_a (E*3)
    unsigned short* Wp6 = (unsigned short*)alloc((size_t)8  * 512 * 256 * 2);     // eW3
    unsigned short* Wp7 = (unsigned short*)alloc((size_t)24 * 256 * 256 * 2);     // eW_b (E*3)
    unsigned short* A0  = (unsigned short*)alloc((size_t)NPTS * 96 * 2);
    unsigned short* hB  = (unsigned short*)alloc((size_t)NPTS * DD * 2);          // reused as hout
    unsigned short* t1  = (unsigned short*)alloc((size_t)NPTS * DD * 2);          // t1+ext reused as A2
    unsigned short* ext = (unsigned short*)alloc((size_t)NPTS * DD * 2);
    int*   eidx = (int*)  alloc((size_t)NPTS * 4);
    float* gate = (float*)alloc((size_t)NPTS * 4);
    int*   loc  = (int*)  alloc((size_t)NPTS * 4);
    int*   bcnt = (int*)  alloc(32 * EE * 4);
    int*   bbas = (int*)  alloc(32 * EE * 4);
    unsigned short* X0 = (unsigned short*)alloc((size_t)EE * CAPC * DD * 2);      // reused as h2
    unsigned short* X1 = (unsigned short*)alloc((size_t)EE * CAPC * DD * 2);
    unsigned short* X2 = (unsigned short*)alloc((size_t)EE * CAPC * DD * 2);
    unsigned short* A2   = t1;   // N x 352 overlays t1+ext (dead by then)
    unsigned short* hout = hB;   // overlays hB (dead after scatter)
    unsigned short* h2   = X0;   // overlays X0 (dead after concat layer)

    const long ROWS = (long)EE * CAPC;        // 163840
    const int  GN   = NPTS / 32;              // 4096
    const int  GM   = (int)(ROWS / 32);       // 5120
    const int  CB   = CAPC / 32;              // 640 row-blocks per expert

    // ---- 1. repack all weights to WMMA fragment order (bf16) ----
    k_pack<<<dim3((96*256+255)/256, 1),  256, 0, stream>>>(w_xyz, Wp0, 75, 96, 256);
    k_pack<<<dim3(256, 1),               256, 0, stream>>>(wg1,   Wp1, 256, 256, 256);
    k_pack<<<dim3(256, 1),               256, 0, stream>>>(wg2,   Wp2, 256, 256, 256);
    k_pack<<<dim3(256, 1),               256, 0, stream>>>(w1,    Wp3, 256, 256, 256);
    k_pack<<<dim3((352*128+255)/256, 1), 256, 0, stream>>>(w2,    Wp4, 331, 352, 128);
    k_pack<<<dim3(256, 24),              256, 0, stream>>>(eW_a,  Wp5, 256, 256, 256);
    k_pack<<<dim3(512, 8),               256, 0, stream>>>(eW3,   Wp6, 512, 512, 256);
    k_pack<<<dim3(256, 24),              256, 0, stream>>>(eW_b,  Wp7, 256, 256, 256);

    // ---- 2. posenc + trunk GEMMs ----
    k_posenc<<<NPTS/256, 256, 0, stream>>>(x, A0);
    k_gemm<4><<<GN, 256, 0, stream>>>(A0, A0, 96,  Wp0, 0, 0, b_xyz, 0, hB,  256,  96, 256, 0);
    k_gemm<4><<<GN, 256, 0, stream>>>(hB, hB, 256, Wp1, 0, 0, bg1,   0, t1,  256, 256, 256, 1);
    k_gemm<4><<<GN, 256, 0, stream>>>(t1, t1, 256, Wp2, 0, 0, bg2,   0, ext, 256, 256, 256, 0);

    // ---- 3. gating + token-order capacity assignment ----
    k_gate<<<NPTS/8, 256, 0, stream>>>(ext, ln_g, ln_b, wg, eidx, gate);
    k_count<<<32, 256, 0, stream>>>(eidx, bcnt);
    k_scan<<<1, 32, 0, stream>>>(bcnt, bbas);
    k_assign<<<32, 256, 0, stream>>>(eidx, bbas, loc);

    // ---- 4. dispatch ----
    k_zero<<<2048, 256, 0, stream>>>((uint4*)X0, ROWS * DD / 8);
    k_scatter<<<NPTS/8, 256, 0, stream>>>(hB, eidx, loc, X0);

    // ---- 5. expert MLP chain (full E x CAP buffer, like the reference einsums) ----
    const long WS = (long)3 * 256 * 256;   // per-expert stride in Wp5/Wp7 (halves)
    k_gemm<4><<<GM, 256, 0, stream>>>(X0, X0, 256, Wp5,            WS, CB, eb_a,        3*DD, X1, 256, 256, 256, 1);
    k_gemm<4><<<GM, 256, 0, stream>>>(X1, X1, 256, Wp5 + 65536,    WS, CB, eb_a + DD,   3*DD, X2, 256, 256, 256, 1);
    k_gemm<4><<<GM, 256, 0, stream>>>(X2, X2, 256, Wp5 + 2*65536,  WS, CB, eb_a + 2*DD, 3*DD, X1, 256, 256, 256, 1);
    k_gemm<4><<<GM, 256, 0, stream>>>(X0, X1, 256, Wp6, (long)512*256, CB, eb3,           DD, X2, 256, 512, 256, 1);
    k_gemm<4><<<GM, 256, 0, stream>>>(X2, X2, 256, Wp7,            WS, CB, eb_b,        3*DD, X1, 256, 256, 256, 1);
    k_gemm<4><<<GM, 256, 0, stream>>>(X1, X1, 256, Wp7 + 65536,    WS, CB, eb_b + DD,   3*DD, X2, 256, 256, 256, 1);
    k_gemm<4><<<GM, 256, 0, stream>>>(X2, X2, 256, Wp7 + 2*65536,  WS, CB, eb_b + 2*DD, 3*DD, X1, 256, 256, 256, 0);

    // ---- 6. combine (gate * keep, relu) + sigma head ----
    k_combine<<<NPTS/8, 256, 0, stream>>>(X1, eidx, loc, gate, wsv, bs, hout, out);

    // ---- 7. color branch: h@w1 -> concat(dir posenc, appearance) -> w2 -> rgb ----
    k_gemm<4><<<GN, 256, 0, stream>>>(hout, hout, 256, Wp3, 0, 0, b1, 0, A2, 352, 256, 256, 0);
    k_tail<<<NPTS/256, 256, 0, stream>>>(x, emb_a, A2);
    k_gemm<2><<<GN, 256, 0, stream>>>(A2, A2, 352, Wp4, 0, 0, b2, 0, h2, 128, 352, 128, 1);
    k_rgb<<<NPTS/8, 256, 0, stream>>>(h2, wcm, bc, out);
}
